// DecoderLayer_30296699306620
// MI455X (gfx1250) — compile-verified, matrix operand order checked by reading, not run on hardware
//
#include <hip/hip_runtime.h>
#include <hip/hip_bf16.h>

// ---------------------------------------------------------------------------
// XLNet relative-attention decoder layer for MI455X (gfx1250, wave32, WMMA).
// B=1, Q=2048, C=2048, H=1024, N=16 heads, D=64, F=4096, R=Q+C=4096.
// All GEMM work goes through v_wmma_f32_16x16x32_f16; f16-A GEMM staging uses
// GLOBAL_LOAD_ASYNC_TO_LDS_B128 when the toolchain exposes the builtin.
// ---------------------------------------------------------------------------

#define QLEN 2048
#define CLEN 2048
#define HDIM 1024
#define NHEAD 16
#define DHEAD 64
#define FDIM 4096
#define RLEN 4096
#define NEGV 1e30f
#define EPSV 1e-12f

typedef __attribute__((ext_vector_type(16))) _Float16 v16h;
typedef __attribute__((ext_vector_type(8)))  _Float16 h8;
typedef __attribute__((ext_vector_type(8)))  float    v8f;
typedef __attribute__((ext_vector_type(4)))  float    f4;
typedef __attribute__((ext_vector_type(4)))  int      i4;

#if __has_builtin(__builtin_amdgcn_global_load_async_to_lds_b128)
#define HAVE_ASYNC 1
#else
#define HAVE_ASYNC 0
#endif

#define AS1 __attribute__((address_space(1)))
#define AS3 __attribute__((address_space(3)))

static __device__ inline void async_copy_b128(const _Float16* g, _Float16* l) {
#if HAVE_ASYNC
  // Builtin signature (from clang diagnostics): (AS1 i4*, AS3 i4*, imm, imm).
  // Low 32 bits of a generic shared pointer are the LDS byte offset
  // (per the aperture rules).
  __builtin_amdgcn_global_load_async_to_lds_b128(
      (AS1 i4*)(uintptr_t)g,
      (AS3 i4*)(uintptr_t)(unsigned)(uintptr_t)l, 0, 0);
#else
  *(h8*)l = *(const h8*)g;
#endif
}

static __device__ inline void wait_async() {
#if HAVE_ASYNC
#if __has_builtin(__builtin_amdgcn_s_wait_asynccnt)
  __builtin_amdgcn_s_wait_asynccnt(0);
#else
  asm volatile("s_wait_asynccnt 0x0" ::: "memory");
#endif
#endif
}

static __device__ inline v8f wmma16(v16h a, v16h b, v8f c) {
  // D = A(16x32 f16) * B(32x16 f16) + C(16x16 f32)
  return __builtin_amdgcn_wmma_f32_16x16x32_f16(false, a, false, b,
                                                (short)0, c, false, false);
}

static __device__ inline v16h pack2(h8 lo, h8 hi) {
  v16h a;
#pragma unroll
  for (int i = 0; i < 8; ++i) { a[i] = lo[i]; a[i + 8] = hi[i]; }
  return a;
}

// A-fragment (16x32, 16-bit): lane holds row = lane&15; K halves are two
// contiguous 8-element runs at off and off+16, off = 8*(lane>=16).
static __device__ inline v16h load_afrag_bias(const _Float16* rowp,
                                              const float* bias64,
                                              int kbase, int lane) {
  int off = kbase + ((lane >> 4) << 3);
  h8 r0 = *(const h8*)(rowp + off);
  h8 r1 = *(const h8*)(rowp + off + 16);
  v16h a;
#pragma unroll
  for (int i = 0; i < 8; ++i) {
    a[i]     = (_Float16)((float)r0[i] + bias64[off + i]);
    a[i + 8] = (_Float16)((float)r1[i] + bias64[off + 16 + i]);
  }
  return a;
}

// B-fragment (32x16, 16-bit): lane holds col = lane&15 (a K/V/R row index).
static __device__ inline v16h load_bfrag_rows(const _Float16* base,
                                              int row, int kbase, int lane) {
  int off = kbase + ((lane >> 4) << 3);
  const _Float16* p = base + (size_t)row * HDIM + off;
  return pack2(*(const h8*)p, *(const h8*)(p + 16));
}

// ---------------------------------------------------------------------------
// Templated WMMA GEMM: out = epilogue(A[MxK] @ B[KxN]).
// Block tile 64x128, 8 waves in a 2x4 grid, each wave owns 32x32
// (4 v_wmma per K-step).  All flags are compile-time -> straight-line code.
//   AF16  : A is f16 (uses async global->LDS staging), else f32 (convert)
//   TRANSB: B element (k,n) = Bf[n*K + k]
// M % 64 == 0, N % 128 == 0, K % 32 == 0 (true for all uses here).
// ---------------------------------------------------------------------------
template <int AF16, int TRANSB, int BIAS, int RELU, int RES, int OUTH, int OUTF>
__global__ __launch_bounds__(256) void gemm_wmma(
    const void* __restrict__ Aptr, const float* __restrict__ Bf,
    int M, int K, int N,
    const float* __restrict__ bias, const float* __restrict__ resid,
    _Float16* __restrict__ outH, float* __restrict__ outF) {
  __shared__ _Float16 Ah[64 * 40];
  __shared__ _Float16 Bt[128 * 40];   // transposed: Bt[n][k]

  const int tid  = threadIdx.x;
  const int lane = tid & 31;
  const int wid  = tid >> 5;
  const int m0 = blockIdx.y * 64;
  const int n0 = blockIdx.x * 128;
  const int wr = (wid & 1) * 32;    // wave row offset in block tile
  const int wc = (wid >> 1) * 32;   // wave col offset in block tile

  const float*    Af  = (const float*)Aptr;
  const _Float16* A16 = (const _Float16*)Aptr;

  v8f c00 = {}; v8f c01 = {}; v8f c10 = {}; v8f c11 = {};

  for (int k0 = 0; k0 < K; k0 += 32) {
    { // ---- stage A tile 64x32 -> f16 LDS ----
      int r = tid >> 2, s = (tid & 3) * 8;
      if (AF16) {
        async_copy_b128(A16 + (size_t)(m0 + r) * K + k0 + s, &Ah[r * 40 + s]);
      } else {
        const float* src = Af + (size_t)(m0 + r) * K + k0 + s;
        f4 x0 = *(const f4*)src;
        f4 x1 = *(const f4*)(src + 4);
        h8 hv;
#pragma unroll
        for (int i = 0; i < 4; ++i) {
          hv[i]     = (_Float16)x0[i];
          hv[i + 4] = (_Float16)x1[i];
        }
        *(h8*)&Ah[r * 40 + s] = hv;
      }
    }
    if (TRANSB) { // ---- B[n][k]: contiguous in k, vector both sides ----
      int n = tid >> 1, s = (tid & 1) * 16;
      const float* src = Bf + (size_t)(n0 + n) * K + k0 + s;
      f4 x0 = *(const f4*)src;
      f4 x1 = *(const f4*)(src + 4);
      f4 x2 = *(const f4*)(src + 8);
      f4 x3 = *(const f4*)(src + 12);
      h8 h0, h1;
#pragma unroll
      for (int i = 0; i < 4; ++i) {
        h0[i] = (_Float16)x0[i]; h0[i + 4] = (_Float16)x1[i];
        h1[i] = (_Float16)x2[i]; h1[i + 4] = (_Float16)x3[i];
      }
      *(h8*)&Bt[n * 40 + s]     = h0;
      *(h8*)&Bt[n * 40 + s + 8] = h1;
    } else {      // ---- B[k][n]: contiguous in n, vector global loads ----
      int k = tid >> 4, nb = (tid & 15) * 8;
#pragma unroll
      for (int kk2 = 0; kk2 < 2; ++kk2) {
        int kk = k + kk2 * 16;
        const float* src = Bf + (size_t)(k0 + kk) * N + n0 + nb;
        f4 x0 = *(const f4*)src;
        f4 x1 = *(const f4*)(src + 4);
#pragma unroll
        for (int i = 0; i < 4; ++i) {
          Bt[(nb + i) * 40 + kk]       = (_Float16)x0[i];
          Bt[(nb + 4 + i) * 40 + kk]   = (_Float16)x1[i];
        }
      }
    }
    if (AF16) wait_async();
    __syncthreads();

    // ---- fragments + 4 WMMAs (32x32 wave tile) ----
    const int off = (lane >> 4) << 3;
    const int j   = lane & 15;
    v16h a0 = pack2(*(const h8*)&Ah[(wr + j) * 40 + off],
                    *(const h8*)&Ah[(wr + j) * 40 + off + 16]);
    v16h a1 = pack2(*(const h8*)&Ah[(wr + 16 + j) * 40 + off],
                    *(const h8*)&Ah[(wr + 16 + j) * 40 + off + 16]);
    v16h b0 = pack2(*(const h8*)&Bt[(wc + j) * 40 + off],
                    *(const h8*)&Bt[(wc + j) * 40 + off + 16]);
    v16h b1 = pack2(*(const h8*)&Bt[(wc + 16 + j) * 40 + off],
                    *(const h8*)&Bt[(wc + 16 + j) * 40 + off + 16]);
    c00 = wmma16(a0, b0, c00);
    c01 = wmma16(a0, b1, c01);
    c10 = wmma16(a1, b0, c10);
    c11 = wmma16(a1, b1, c11);
    __syncthreads();
  }

  // ---- epilogue: C/D layout -> row = v + 8*(lane>=16), col = lane&15 ----
  const int lh = (lane & 16) ? 8 : 0;
  const int j  = lane & 15;
#pragma unroll
  for (int rg = 0; rg < 2; ++rg) {
#pragma unroll
    for (int cg = 0; cg < 2; ++cg) {
      v8f acc = (rg == 0) ? (cg == 0 ? c00 : c01) : (cg == 0 ? c10 : c11);
#pragma unroll
      for (int v = 0; v < 8; ++v) {
        int row = m0 + wr + rg * 16 + v + lh;
        int col = n0 + wc + cg * 16 + j;
        float val = acc[v];
        if (BIAS) val += bias[col];
        if (RELU) val = fmaxf(val, 0.0f);
        if (RES)  val += resid[(size_t)row * N + col];
        size_t idx = (size_t)row * N + col;
        if (OUTH) outH[idx] = (_Float16)val;
        if (OUTF) outF[idx] = val;
      }
    }
  }
}

// ---------------------------------------------------------------------------
// Fused relative attention (flash-style online softmax).
// grid = (Q/128, NHEAD); block = 256 (8 waves, 16 query rows per wave).
// bd[q,c] = (q+pb).r[Q+c-q]  -> 3 WMMA window tiles + shuffle extraction.
// ---------------------------------------------------------------------------
__global__ __launch_bounds__(256) void attn_relshift_wmma(
    const _Float16* __restrict__ qh, const _Float16* __restrict__ kh,
    const _Float16* __restrict__ vh, const _Float16* __restrict__ rh,
    const float* __restrict__ cbias, const float* __restrict__ pbias,
    const float* __restrict__ sbias, const float* __restrict__ segenc,
    const int* __restrict__ segmat, const float* __restrict__ mask,
    _Float16* __restrict__ attnout) {
  __shared__ _Float16 Vt[64 * 32];     // V chunk transposed: Vt[d][c]
  __shared__ _Float16 Pl[8][16 * 32];  // per-wave prob tile [row][c]

  const int tid  = threadIdx.x;
  const int lane = tid & 31;
  const int wid  = tid >> 5;
  const int head = blockIdx.y;
  const int q0   = blockIdx.x * 128 + wid * 16;
  const int j    = lane & 15;
  const int lh   = (lane & 16) ? 8 : 0;

  const _Float16* kbasep = kh + head * DHEAD;
  const _Float16* rbasep = rh + head * DHEAD;

  // ----- per-row segment-bias dot products: e0/e1 for row = lane&15 -----
  float e0 = 0.0f, e1 = 0.0f;
  {
    int row = j, dbase = (lane >> 4) * 32;
    const _Float16* qp_ = qh + (size_t)(q0 + row) * HDIM + head * DHEAD;
    const float* sb  = sbias + head * DHEAD;
    const float* sg0 = segenc + head * DHEAD;
    const float* sg1 = segenc + NHEAD * DHEAD + head * DHEAD;
#pragma unroll 8
    for (int d = 0; d < 32; ++d) {
      float qv = (float)qp_[dbase + d] + sb[dbase + d];
      e0 += qv * sg0[dbase + d];
      e1 += qv * sg1[dbase + d];
    }
    e0 += __shfl_xor(e0, 16, 32);
    e1 += __shfl_xor(e1, 16, 32);
  }

  // ----- Q fragments (held in registers for the whole C loop) -----
  const _Float16* qrow = qh + (size_t)(q0 + j) * HDIM + head * DHEAD;
  v16h qc0 = load_afrag_bias(qrow, cbias + head * DHEAD, 0,  lane);
  v16h qc1 = load_afrag_bias(qrow, cbias + head * DHEAD, 32, lane);
  v16h qp0 = load_afrag_bias(qrow, pbias + head * DHEAD, 0,  lane);
  v16h qp1 = load_afrag_bias(qrow, pbias + head * DHEAD, 32, lane);

  v8f acc[4]; acc[0] = {}; acc[1] = {}; acc[2] = {}; acc[3] = {};
  float mrow[8], lsum[8];
#pragma unroll
  for (int v = 0; v < 8; ++v) { mrow[v] = -NEGV; lsum[v] = 0.0f; }

  _Float16* pl = &Pl[wid][0];

  for (int c0 = 0; c0 < CLEN; c0 += 32) {
    // stage V chunk transposed (shared by all 8 waves)
    {
      int d = tid >> 2, cs = (tid & 3) * 8;
#pragma unroll
      for (int t = 0; t < 8; ++t) {
        int c = cs + t;
        Vt[d * 32 + c] = vh[(size_t)(c0 + c) * HDIM + head * DHEAD + d];
      }
    }
    __syncthreads();

    // ----- AC = (q + cb) . k^T, two 16-col subtiles -----
    v8f s0 = {}; v8f s1 = {};
    {
      int cr = c0 + j;
      s0 = wmma16(qc0, load_bfrag_rows(kbasep, cr,       0,  lane), s0);
      s0 = wmma16(qc1, load_bfrag_rows(kbasep, cr,       32, lane), s0);
      s1 = wmma16(qc0, load_bfrag_rows(kbasep, cr + 16,  0,  lane), s1);
      s1 = wmma16(qc1, load_bfrag_rows(kbasep, cr + 16,  32, lane), s1);
    }

    // ----- BD window tiles: W0/W1/W2 cover r rows rbase .. rbase+47 -----
    int rbase = QLEN + c0 - q0 - 15;
    v8f w0 = {}; v8f w1 = {}; v8f w2 = {};
    {
      int r0i = rbase + j;       r0i = r0i < 0 ? 0 : (r0i > RLEN - 1 ? RLEN - 1 : r0i);
      int r1i = rbase + 16 + j;  r1i = r1i < 0 ? 0 : (r1i > RLEN - 1 ? RLEN - 1 : r1i);
      int r2i = rbase + 32 + j;  r2i = r2i < 0 ? 0 : (r2i > RLEN - 1 ? RLEN - 1 : r2i);
      w0 = wmma16(qp0, load_bfrag_rows(rbasep, r0i, 0,  lane), w0);
      w0 = wmma16(qp1, load_bfrag_rows(rbasep, r0i, 32, lane), w0);
      w1 = wmma16(qp0, load_bfrag_rows(rbasep, r1i, 0,  lane), w1);
      w1 = wmma16(qp1, load_bfrag_rows(rbasep, r1i, 32, lane), w1);
      w2 = wmma16(qp0, load_bfrag_rows(rbasep, r2i, 0,  lane), w2);
      w2 = wmma16(qp1, load_bfrag_rows(rbasep, r2i, 32, lane), w2);
    }

    // ----- rel-shift extraction + ef + scale + mask -----
#pragma unroll
    for (int v = 0; v < 8; ++v) {
      int i  = v + lh;                 // query row within tile
      int o  = 15 + j - i;             // window offset, in [0,30]
      int sl = (o & 15) | (lane & 16); // source lane in same half
      float f0 = __shfl(w0[v], sl, 32);
      float f1 = __shfl(w1[v], sl, 32);
      float f2 = __shfl(w2[v], sl, 32);
      s0[v] += (o < 16) ? f0 : f1;     // subtile0 uses W0/W1
      s1[v] += (o < 16) ? f1 : f2;     // subtile1 uses W1/W2

      float er0 = __shfl(e0, i, 32);
      float er1 = __shfl(e1, i, 32);
      int qrowg = q0 + i;
      int col0  = c0 + j;
      int sm0 = segmat[(size_t)qrowg * CLEN + col0];
      int sm1 = segmat[(size_t)qrowg * CLEN + col0 + 16];
      s0[v] = (s0[v] + (sm0 ? er1 : er0)) * 0.125f
              - NEGV * mask[(size_t)qrowg * CLEN + col0];
      s1[v] = (s1[v] + (sm1 ? er1 : er0)) * 0.125f
              - NEGV * mask[(size_t)qrowg * CLEN + col0 + 16];
    }

    // ----- online softmax over this 32-column chunk -----
#pragma unroll
    for (int v = 0; v < 8; ++v) {
      float t = fmaxf(s0[v], s1[v]);
      t = fmaxf(t, __shfl_xor(t, 1, 32));
      t = fmaxf(t, __shfl_xor(t, 2, 32));
      t = fmaxf(t, __shfl_xor(t, 4, 32));
      t = fmaxf(t, __shfl_xor(t, 8, 32));
      float mn = fmaxf(mrow[v], t);
      float sc = __expf(mrow[v] - mn);
      mrow[v] = mn;
      float p0 = __expf(s0[v] - mn);
      float p1 = __expf(s1[v] - mn);
      float rs = p0 + p1;
      rs += __shfl_xor(rs, 1, 32);
      rs += __shfl_xor(rs, 2, 32);
      rs += __shfl_xor(rs, 4, 32);
      rs += __shfl_xor(rs, 8, 32);
      lsum[v] = lsum[v] * sc + rs;
      s0[v] = p0; s1[v] = p1;
#pragma unroll
      for (int t4 = 0; t4 < 4; ++t4) acc[t4][v] *= sc;
    }

    // ----- probs -> per-wave LDS, reload as A fragment, P.V -----
#pragma unroll
    for (int v = 0; v < 8; ++v) {
      int i = v + lh;
      pl[i * 32 + j]      = (_Float16)s0[v];
      pl[i * 32 + 16 + j] = (_Float16)s1[v];
    }
    {
      int off = (lane >> 4) << 3;
      v16h pa = pack2(*(const h8*)(pl + j * 32 + off),
                      *(const h8*)(pl + j * 32 + off + 16));
#pragma unroll
      for (int t = 0; t < 4; ++t) {
        int d = t * 16 + j;
        v16h vb = pack2(*(const h8*)(&Vt[d * 32 + off]),
                        *(const h8*)(&Vt[d * 32 + off + 16]));
        acc[t] = wmma16(pa, vb, acc[t]);
      }
    }
    __syncthreads();  // protect Vt before next chunk staging
  }

  // ----- normalize and store attn (f16, layout [Q, N*D]) -----
#pragma unroll
  for (int v = 0; v < 8; ++v) {
    float inv = 1.0f / lsum[v];
    int row = q0 + v + lh;
#pragma unroll
    for (int t = 0; t < 4; ++t) {
      int col = head * DHEAD + t * 16 + j;
      attnout[(size_t)row * HDIM + col] = (_Float16)(acc[t][v] * inv);
    }
  }
}

// ---------------------------------------------------------------------------
// Row LayerNorm: one row per block, 256 threads.  Writes f32 and optional f16.
// ---------------------------------------------------------------------------
__global__ __launch_bounds__(256) void layernorm_row(
    const float* __restrict__ x, const float* __restrict__ g,
    const float* __restrict__ b, float* __restrict__ outF,
    _Float16* __restrict__ outH, int ncols) {
  __shared__ float red[256];
  const int row = blockIdx.x;
  const int tid = threadIdx.x;
  const float* xr = x + (size_t)row * ncols;

  float s = 0.0f;
  for (int c = tid; c < ncols; c += 256) s += xr[c];
  red[tid] = s; __syncthreads();
  for (int o = 128; o > 0; o >>= 1) {
    if (tid < o) red[tid] += red[tid + o];
    __syncthreads();
  }
  float mean = red[0] / ncols;
  __syncthreads();

  float vs = 0.0f;
  for (int c = tid; c < ncols; c += 256) {
    float d = xr[c] - mean; vs += d * d;
  }
  red[tid] = vs; __syncthreads();
  for (int o = 128; o > 0; o >>= 1) {
    if (tid < o) red[tid] += red[tid + o];
    __syncthreads();
  }
  float rstd = rsqrtf(red[0] / ncols + EPSV);

  for (int c = tid; c < ncols; c += 256) {
    float y = (xr[c] - mean) * rstd * g[c] + b[c];
    size_t idx = (size_t)row * ncols + c;
    if (outF) outF[idx] = y;
    if (outH) outH[idx] = (_Float16)y;
  }
}

// ---------------------------------------------------------------------------
extern "C" void kernel_launch(void* const* d_in, const int* in_sizes, int n_in,
                              void* d_out, int out_size, void* d_ws,
                              size_t ws_size, hipStream_t stream) {
  const float* content_stream = (const float*)d_in[0];
  const float* content_mask   = (const float*)d_in[1];
  const float* context        = (const float*)d_in[2];
  const float* pos_enc        = (const float*)d_in[3];
  const float* content_bias   = (const float*)d_in[4];
  const float* position_bias  = (const float*)d_in[5];
  const float* seg_enc        = (const float*)d_in[6];
  const int*   seg_mat        = (const int*)d_in[7];
  const float* segment_bias   = (const float*)d_in[8];
  const float* Wq = (const float*)d_in[9];
  const float* Wk = (const float*)d_in[10];
  const float* Wv = (const float*)d_in[11];
  const float* Wr = (const float*)d_in[12];
  const float* Wo = (const float*)d_in[13];
  const float* ln1g = (const float*)d_in[14];
  const float* ln1b = (const float*)d_in[15];
  const float* W1 = (const float*)d_in[16];
  const float* b1 = (const float*)d_in[17];
  const float* W2 = (const float*)d_in[18];
  const float* b2 = (const float*)d_in[19];
  const float* ln2g = (const float*)d_in[20];
  const float* ln2b = (const float*)d_in[21];
  float* out = (float*)d_out;

  // workspace carve (~68 MB), 256B aligned
  size_t off = 0;
  auto carve = [&](size_t bytes) -> void* {
    void* p = (char*)d_ws + off;
    off += (bytes + 255) & ~(size_t)255;
    return p;
  };
  _Float16* qf16   = (_Float16*)carve((size_t)QLEN * HDIM * 2);
  _Float16* kf16   = (_Float16*)carve((size_t)CLEN * HDIM * 2);
  _Float16* vf16   = (_Float16*)carve((size_t)CLEN * HDIM * 2);
  _Float16* rf16   = (_Float16*)carve((size_t)RLEN * HDIM * 2);
  _Float16* attn16 = (_Float16*)carve((size_t)QLEN * HDIM * 2);
  float*    attres = (float*)carve((size_t)QLEN * HDIM * 4);
  float*    ffninF = (float*)carve((size_t)QLEN * HDIM * 4);
  _Float16* ffninH = (_Float16*)carve((size_t)QLEN * HDIM * 2);
  _Float16* h1f16  = (_Float16*)carve((size_t)QLEN * FDIM * 2);
  float*    ffnsum = (float*)carve((size_t)QLEN * HDIM * 4);

  dim3 blk(256);

  // 1) projections: q,k,v,r (f32 in -> f16 out)   [AF16,TRANSB,BIAS,RELU,RES,OUTH,OUTF]
  gemm_wmma<0,0,0,0,0,1,0><<<dim3(HDIM / 128, QLEN / 64), blk, 0, stream>>>(
      content_stream, Wq, QLEN, HDIM, HDIM, nullptr, nullptr, qf16, nullptr);
  gemm_wmma<0,0,0,0,0,1,0><<<dim3(HDIM / 128, CLEN / 64), blk, 0, stream>>>(
      context, Wk, CLEN, HDIM, HDIM, nullptr, nullptr, kf16, nullptr);
  gemm_wmma<0,0,0,0,0,1,0><<<dim3(HDIM / 128, CLEN / 64), blk, 0, stream>>>(
      context, Wv, CLEN, HDIM, HDIM, nullptr, nullptr, vf16, nullptr);
  gemm_wmma<0,0,0,0,0,1,0><<<dim3(HDIM / 128, RLEN / 64), blk, 0, stream>>>(
      pos_enc, Wr, RLEN, HDIM, HDIM, nullptr, nullptr, rf16, nullptr);

  // 2) fused relative attention
  attn_relshift_wmma<<<dim3(QLEN / 128, NHEAD), blk, 0, stream>>>(
      qf16, kf16, vf16, rf16, content_bias, position_bias, segment_bias,
      seg_enc, seg_mat, content_mask, attn16);

  // 3) output projection (B = Wo^T) + residual
  gemm_wmma<1,1,0,0,1,0,1><<<dim3(HDIM / 128, QLEN / 64), blk, 0, stream>>>(
      attn16, Wo, QLEN, HDIM, HDIM, nullptr, content_stream, nullptr, attres);

  // 4) LN1 -> ffn_in (f32 + f16)
  layernorm_row<<<dim3(QLEN), blk, 0, stream>>>(attres, ln1g, ln1b,
                                                ffninF, ffninH, HDIM);

  // 5) FFN up: relu(ffn_in @ W1 + b1) -> f16
  gemm_wmma<1,0,1,1,0,1,0><<<dim3(FDIM / 128, QLEN / 64), blk, 0, stream>>>(
      ffninH, W1, QLEN, HDIM, FDIM, b1, nullptr, h1f16, nullptr);

  // 6) FFN down: h1 @ W2 + b2 + ffn_in -> f32
  gemm_wmma<1,0,1,0,1,0,1><<<dim3(HDIM / 128, QLEN / 64), blk, 0, stream>>>(
      h1f16, W2, QLEN, FDIM, HDIM, b2, ffninF, nullptr, ffnsum);

  // 7) LN2 -> output
  layernorm_row<<<dim3(QLEN), blk, 0, stream>>>(ffnsum, ln2g, ln2b,
                                                out, nullptr, HDIM);
}
